// AwkwardDeepSetDoubleJagged_24979529794079
// MI455X (gfx1250) — matrix operand order repaired and checked by reading, not compiled, not run on hardware
//
#include <hip/hip_runtime.h>

// DeepSets (double-jagged) fused pipeline for gfx1250 / MI455X.
//  K1: zero workspace
//  K2: fused phi1 (1->64->64) + sorted-segment-sum via run-merged f32 atomics  [WMMA bf16]
//  K3: 5-layer 64x64 event MLP chain, weights in LDS, per-wave act restage     [WMMA bf16]
//  K4: tiny rho2 + log_softmax tail (1 block, VALU)

#define H 64
#define EVENTS 20000
#define NCLS 10
#define LSTR 66   // padded bf16 row stride (in halves) to break LDS bank alignment

typedef __bf16 v16bf __attribute__((ext_vector_type(16)));
typedef float  v8f   __attribute__((ext_vector_type(8)));

union Frag { v16bf v; unsigned int u[8]; };
union Pk2  { __bf16 h[2]; unsigned int u; };

static __device__ __forceinline__ v8f wmma_bf16(const Frag& a, const Frag& b, v8f c) {
  // D = A(16x32 bf16) * B(32x16 bf16) + C(16x16 f32)
  return __builtin_amdgcn_wmma_f32_16x16x32_bf16(false, a.v, false, b.v, (short)0, c,
                                                 false, false);
}

static __device__ __forceinline__ void lds_fence() {
  // LDS is in-order per wave on CDNA5; this only pins compiler scheduling.
  asm volatile("" ::: "memory");
  __builtin_amdgcn_wave_barrier();
  asm volatile("" ::: "memory");
}

__global__ void k_zero(float* __restrict__ p, int n) {
  int i = blockIdx.x * blockDim.x + threadIdx.x;
  if (i < n) p[i] = 0.0f;
}

// ---------------------------------------------------------------------------
// Kernel A: per-particle phi1 + segment sum.  One wave owns a 16-particle tile.
// ---------------------------------------------------------------------------
__global__ __launch_bounds__(256) void k_phi1_segsum(
    const float* __restrict__ values, const int* __restrict__ segids,
    const float* __restrict__ p1w0, const float* __restrict__ p1b0,
    const float* __restrict__ p1w1, const float* __restrict__ p1b1,
    float* __restrict__ ev, int n, int ntiles, int nwaves)
{
  __shared__ __bf16 s_w1[H * LSTR];       // [n_out][k] bf16 (torch layout, B-friendly)
  __shared__ float  s_w0[H], s_b0[H], s_b1[H];
  __shared__ __bf16 s_h0[8][16 * LSTR];   // per-wave A staging [m][k]
  __shared__ float  s_v[8][16];
  __shared__ int    s_sg[8][16];

  const int tid = threadIdx.x;
  for (int e = tid; e < H * H; e += 256)
    s_w1[(e >> 6) * LSTR + (e & 63)] = (__bf16)p1w1[e];
  if (tid < H) { s_w0[tid] = p1w0[tid]; s_b0[tid] = p1b0[tid]; s_b1[tid] = p1b1[tid]; }
  __syncthreads();

  const int lane = tid & 31, wave = tid >> 5;
  const int hi = lane >> 4, ml = lane & 15;
  const int m0 = lane >> 1, koff = (lane & 1) * 32;   // h0 compute assignment

  // All 8 B fragments (the 64x64 weight) stay resident in VGPRs for the kernel.
  Frag Bf[2][4];
  for (int kh = 0; kh < 2; ++kh)
    for (int nt = 0; nt < 4; ++nt) {
      const int nc = nt * 16 + ml;
      for (int p = 0; p < 8; ++p) {
        const int k = kh * 32 + 2 * p + ((p < 4) ? 0 : 8) + hi * 8;
        Bf[kh][nt].u[p] = *(const unsigned int*)&s_w1[nc * LSTR + k];
      }
    }

  // Per-lane packed (w0,b0) pairs for its fixed k-range (constant over tiles).
  unsigned int pw[16], pb[16];
  for (int i = 0; i < 16; ++i) {
    Pk2 a, b;
    a.h[0] = (__bf16)s_w0[koff + 2 * i]; a.h[1] = (__bf16)s_w0[koff + 2 * i + 1];
    b.h[0] = (__bf16)s_b0[koff + 2 * i]; b.h[1] = (__bf16)s_b0[koff + 2 * i + 1];
    pw[i] = a.u; pb[i] = b.u;
  }
  float bn[4];
  for (int nt = 0; nt < 4; ++nt) bn[nt] = s_b1[nt * 16 + ml];

  const v8f zero = {0.f, 0.f, 0.f, 0.f, 0.f, 0.f, 0.f, 0.f};
  const int gw = blockIdx.x * 8 + wave;

  for (int t = gw; t < ntiles; t += nwaves) {
    const int base = t * 16;
    if (lane < 16) {
      const int idx = base + lane;
      s_v[wave][lane]  = (idx < n) ? values[idx] : 0.0f;
      s_sg[wave][lane] = (idx < n) ? segids[idx] : -1;
    }
    lds_fence();

    // phi1 layer 0: h0[m][k] = relu(v[m]*w0[k] + b0[k]); 2 lanes per row, 32 k each.
    const float v = s_v[wave][m0];
    for (int i = 0; i < 16; ++i) {
      Pk2 a, b, o;
      a.u = pw[i]; b.u = pb[i];
      const float x0 = fmaxf(0.0f, fmaf(v, (float)a.h[0], (float)b.h[0]));
      const float x1 = fmaxf(0.0f, fmaf(v, (float)a.h[1], (float)b.h[1]));
      o.h[0] = (__bf16)x0; o.h[1] = (__bf16)x1;
      *(unsigned int*)&s_h0[wave][m0 * LSTR + koff + 2 * i] = o.u;
    }
    lds_fence();

    // Gather A fragments (16x32 bf16 x2 k-halves) from LDS.
    Frag Af[2];
    for (int kh = 0; kh < 2; ++kh)
      for (int p = 0; p < 8; ++p) {
        const int kp = 2 * p + ((p < 4) ? 0 : 8) + hi * 8;
        Af[kh].u[p] = *(const unsigned int*)&s_h0[wave][ml * LSTR + kh * 32 + kp];
      }

    // phi1 layer 1: 8 WMMAs -> 16 particles x 64 features, f32 accum.
    v8f acc[4];
    for (int nt = 0; nt < 4; ++nt) {
      acc[nt] = wmma_bf16(Af[0], Bf[0][nt], zero);
      acc[nt] = wmma_bf16(Af[1], Bf[1][nt], acc[nt]);
    }

    // bias + relu + sorted-segment run-merged atomics.
    for (int nt = 0; nt < 4; ++nt) {
      const int nc = nt * 16 + ml;
      float run = 0.0f; int rs = -1;
      for (int r = 0; r < 8; ++r) {
        const int m = r + hi * 8;
        const int sg = s_sg[wave][m];
        const float val = fmaxf(0.0f, acc[nt][r] + bn[nt]);
        if (sg != rs) {
          if (rs >= 0) atomicAdd(&ev[rs * H + nc], run);
          rs = sg; run = 0.0f;
        }
        if (sg >= 0) run += val;
      }
      if (rs >= 0) atomicAdd(&ev[rs * H + nc], run);
    }
  }
}

// ---------------------------------------------------------------------------
// Kernel B: 5 chained 64x64 relu layers over events + global event-sum.
// ---------------------------------------------------------------------------
__global__ __launch_bounds__(256) void k_event_chain(
    const float* __restrict__ ev,
    const float* __restrict__ w0, const float* __restrict__ b0,
    const float* __restrict__ w1, const float* __restrict__ b1,
    const float* __restrict__ w2, const float* __restrict__ b2,
    const float* __restrict__ w3, const float* __restrict__ b3,
    const float* __restrict__ w4, const float* __restrict__ b4,
    float* __restrict__ s2, int nev, int ntiles, int nwaves)
{
  __shared__ __bf16 s_w[5][H * LSTR];
  __shared__ float  s_b[5][H];
  __shared__ __bf16 s_act[8][16 * LSTR];  // per-wave C->A restage
  __shared__ float  s_sum[H];

  const int tid = threadIdx.x;
  const float* W[5] = {w0, w1, w2, w3, w4};
  const float* B[5] = {b0, b1, b2, b3, b4};
  for (int l = 0; l < 5; ++l) {
    for (int e = tid; e < H * H; e += 256)
      s_w[l][(e >> 6) * LSTR + (e & 63)] = (__bf16)W[l][e];
    if (tid < H) s_b[l][tid] = B[l][tid];
  }
  if (tid < H) s_sum[tid] = 0.0f;
  __syncthreads();

  const int lane = tid & 31, wave = tid >> 5;
  const int hi = lane >> 4, ml = lane & 15;
  const int gw = blockIdx.x * 8 + wave;
  const v8f zero = {0.f, 0.f, 0.f, 0.f, 0.f, 0.f, 0.f, 0.f};

  for (int t = gw; t < ntiles; t += nwaves) {
    const int base = t * 16;
    const int row  = base + ml;
    const bool vr  = row < nev;
    const float* rp = ev + (vr ? row : 0) * H;

    // A fragments straight from global ev rows (f32 -> bf16).
    Frag Af[2];
    for (int kh = 0; kh < 2; ++kh)
      for (int p = 0; p < 8; ++p) {
        const int k = kh * 32 + 2 * p + ((p < 4) ? 0 : 8) + hi * 8;
        Pk2 o;
        if (vr) { o.h[0] = (__bf16)rp[k]; o.h[1] = (__bf16)rp[k + 1]; }
        else    { o.u = 0u; }
        Af[kh].u[p] = o.u;
      }

    for (int l = 0; l < 5; ++l) {
      v8f acc[4];
      for (int nt = 0; nt < 4; ++nt) {
        const int nc = nt * 16 + ml;
        Frag Bf0, Bf1;
        for (int p = 0; p < 8; ++p) {
          const int kk = 2 * p + ((p < 4) ? 0 : 8) + hi * 8;
          Bf0.u[p] = *(const unsigned int*)&s_w[l][nc * LSTR + kk];
          Bf1.u[p] = *(const unsigned int*)&s_w[l][nc * LSTR + 32 + kk];
        }
        acc[nt] = wmma_bf16(Af[0], Bf0, zero);
        acc[nt] = wmma_bf16(Af[1], Bf1, acc[nt]);
      }
      if (l < 4) {
        // bias+relu, restage C layout -> A layout through per-wave LDS tile.
        for (int nt = 0; nt < 4; ++nt) {
          const int nc = nt * 16 + ml;
          const float bias = s_b[l][nc];
          for (int r = 0; r < 8; ++r) {
            const int m = r + hi * 8;
            s_act[wave][m * LSTR + nc] = (__bf16)fmaxf(0.0f, acc[nt][r] + bias);
          }
        }
        lds_fence();
        for (int kh = 0; kh < 2; ++kh)
          for (int p = 0; p < 8; ++p) {
            const int kp = 2 * p + ((p < 4) ? 0 : 8) + hi * 8;
            Af[kh].u[p] = *(const unsigned int*)&s_act[wave][ml * LSTR + kh * 32 + kp];
          }
        lds_fence();
      } else {
        // final layer: masked row-sum into block-local LDS accumulator (ds_add_f32).
        for (int nt = 0; nt < 4; ++nt) {
          const int nc = nt * 16 + ml;
          const float bias = s_b[l][nc];
          float part = 0.0f;
          for (int r = 0; r < 8; ++r) {
            const int m = r + hi * 8;
            if (base + m < nev) part += fmaxf(0.0f, acc[nt][r] + bias);
          }
          atomicAdd(&s_sum[nc], part);
        }
      }
    }
  }
  __syncthreads();
  if (tid < H) atomicAdd(&s2[tid], s_sum[tid]);
}

// ---------------------------------------------------------------------------
// Kernel C: rho2 tail + log_softmax (tiny; one 64-thread block).
// ---------------------------------------------------------------------------
__global__ __launch_bounds__(64) void k_finalize(
    const float* __restrict__ s2,
    const float* __restrict__ w0, const float* __restrict__ b0,
    const float* __restrict__ w1, const float* __restrict__ b1,
    const float* __restrict__ ow, const float* __restrict__ ob,
    float* __restrict__ out)
{
  __shared__ float a[H], h1[H], h2[H], o[NCLS];
  const int t = threadIdx.x;
  a[t] = s2[t];
  __syncthreads();
  float acc = b0[t];
  for (int k = 0; k < H; ++k) acc = fmaf(w0[t * H + k], a[k], acc);
  h1[t] = fmaxf(acc, 0.0f);
  __syncthreads();
  acc = b1[t];
  for (int k = 0; k < H; ++k) acc = fmaf(w1[t * H + k], h1[k], acc);
  h2[t] = fmaxf(acc, 0.0f);
  __syncthreads();
  if (t < NCLS) {
    acc = ob[t];
    for (int k = 0; k < H; ++k) acc = fmaf(ow[t * H + k], h2[k], acc);
    o[t] = acc;
  }
  __syncthreads();
  if (t == 0) {
    float mx = o[0];
    for (int j = 1; j < NCLS; ++j) mx = fmaxf(mx, o[j]);
    float s = 0.0f;
    for (int j = 0; j < NCLS; ++j) s += __expf(o[j] - mx);
    const float lse = mx + __logf(s);
    for (int j = 0; j < NCLS; ++j) out[j] = o[j] - lse;
  }
}

extern "C" void kernel_launch(void* const* d_in, const int* in_sizes, int n_in,
                              void* d_out, int out_size, void* d_ws, size_t ws_size,
                              hipStream_t stream) {
  const float* values = (const float*)d_in[0];
  const int*   segids = (const int*)d_in[1];
  const float* p1w0 = (const float*)d_in[2];
  const float* p1b0 = (const float*)d_in[3];
  const float* p1w1 = (const float*)d_in[4];
  const float* p1b1 = (const float*)d_in[5];
  const float* r1w0 = (const float*)d_in[6];
  const float* r1b0 = (const float*)d_in[7];
  const float* r1w1 = (const float*)d_in[8];
  const float* r1b1 = (const float*)d_in[9];
  const float* o1w  = (const float*)d_in[10];
  const float* o1b  = (const float*)d_in[11];
  const float* p2w0 = (const float*)d_in[12];
  const float* p2b0 = (const float*)d_in[13];
  const float* p2w1 = (const float*)d_in[14];
  const float* p2b1 = (const float*)d_in[15];
  const float* r2w0 = (const float*)d_in[16];
  const float* r2b0 = (const float*)d_in[17];
  const float* r2w1 = (const float*)d_in[18];
  const float* r2b1 = (const float*)d_in[19];
  const float* o2w  = (const float*)d_in[20];
  const float* o2b  = (const float*)d_in[21];

  const int n = in_sizes[0];
  float* ev = (float*)d_ws;                 // [EVENTS, 64]
  float* s2 = ev + (size_t)EVENTS * H;      // [64]
  const int zn = EVENTS * H + H;

  k_zero<<<(zn + 255) / 256, 256, 0, stream>>>(ev, zn);

  const int t1 = (n + 15) / 16;
  const int blocksA = 600;                  // 4800 waves, ~26 tiles each
  k_phi1_segsum<<<blocksA, 256, 0, stream>>>(values, segids, p1w0, p1b0, p1w1, p1b1,
                                             ev, n, t1, blocksA * 8);

  const int t2 = (EVENTS + 15) / 16;
  const int blocksB = 80;                   // 640 waves over 1250 tiles
  k_event_chain<<<blocksB, 256, 0, stream>>>(ev, r1w0, r1b0, r1w1, r1b1, o1w, o1b,
                                             p2w0, p2b0, p2w1, p2b1,
                                             s2, EVENTS, t2, blocksB * 8);

  k_finalize<<<1, 64, 0, stream>>>(s2, r2w0, r2b0, r2w1, r2b1, o2w, o2b, (float*)d_out);
}
